// DrugGCN_51745765983004
// MI455X (gfx1250) — matrix operand order repaired and checked by reading, not compiled, not run on hardware
//
#include <hip/hip_runtime.h>

// ---------------------------------------------------------------------------
// DrugGCN forward for MI455X (gfx1250, wave32).
//  - GEMMs: v_wmma_f32_16x16x32_bf16, f32->bf16 packed in LDS, f32 accumulate.
//    A tile kept row-major packed-pair; B tile stored TRANSPOSED in LDS so
//    both fragments load as 16B-aligned uint4 chunks (ds_load_b128 class).
//    Guarded (K=74) path uses clamped-address loads + select-zero so all
//    staging loads pipeline (no exec-masked per-load waits).
//  - Message passing: one wave32 per edge, float4 per lane (512B coalesced
//    gather), 4x global_atomic_add_f32 per lane.
//  - BN: per-column sums via LDS reduce + global atomics; normalization fused
//    into the residual-GEMM epilogue.
// ---------------------------------------------------------------------------

typedef __attribute__((ext_vector_type(16))) __bf16 v16bf;
typedef __attribute__((ext_vector_type(8)))  float  v8f;

#define N_NODES 131072
#define DFEAT   128
#define EPSBN   1e-5f

__device__ __forceinline__ unsigned pack_bf16(float x, float y) {
    union { __bf16 h[2]; unsigned u; } p;     // -> v_cvt_pk_bf16_f32
    p.h[0] = (__bf16)x; p.h[1] = (__bf16)y;
    return p.u;
}

union FragBF { uint4 q[2]; v16bf v; };

// ---------------------------------------------------------------------------
// WMMA GEMM: C[N x 128] = A[N x K] @ B[K x 128]  (A,B f32 row-major)
// Block: 256 threads = 8 waves; block tile = 16 rows x 128 cols,
// wave w computes the 16x16 tile at cols [16w, 16w+16).
// GUARD: K-bounds handling (only the K=74 init GEMM). EPI: fused
//        relu((conv-mean)*invstd*gamma+beta) + (A@B + bias) epilogue.
// ---------------------------------------------------------------------------
template <bool GUARD, bool EPI>
__global__ __launch_bounds__(256)
void gemm_bf16_wmma(const float* A, const float* B, float* C, int K,
                    const float* __restrict__ conv, const float* __restrict__ mean,
                    const float* __restrict__ invstd, const float* __restrict__ gmm,
                    const float* __restrict__ bta, const float* __restrict__ bias)
{
    // Packed bf16 pairs; row stride 20 uints = 80B: 16B aligned, and the
    // 16-lane column reads hit 16 distinct banks (gcd(20,64)=4).
    __shared__ unsigned As_u[16][20];    // A tile 16x32 (pairs along K)
    __shared__ unsigned Bt_u[128][20];   // B tile transposed: [col][K-pair]

    const int tid  = threadIdx.x;
    const int lane = tid & 31;
    const int wave = tid >> 5;        // 0..7 -> output column tile
    const int hi   = lane >> 4;       // lane half (K-half / M-half select)
    const int l15  = lane & 15;
    const int row0 = blockIdx.x * 16;
    const int col0 = wave * 16;

    // Staging coordinates (fixed per thread)
    const int ar  = tid >> 4;         // A row 0..15
    const int ac  = (tid & 15) * 2;   // A col pair base 0..30
    const int bc4 = tid & 31;         // B float4 column (cols 4*bc4..4*bc4+3)
    const int brb = (tid >> 5) * 4;   // B row base: 4 consecutive K rows

    v8f acc = {};
    const int ksteps = (K + 31) >> 5;

    for (int kt = 0; kt < ksteps; ++kt) {
        const int k0 = kt << 5;
        __syncthreads();              // protect LDS reuse across iterations

        // ---- Phase 1: issue ALL global loads unconditionally (clamped
        //      addresses for the padded K=74 path) so they pipeline.
        float2 fa2;
        float4 fb[4];
        bool va = true, vb[4];
        {
            int gk = k0 + ac;         // K even => pair fully valid or invalid
            if (GUARD) { va = (gk < K); if (!va) gk = 0; }
            fa2 = *(const float2*)(A + (size_t)(row0 + ar) * K + gk);
        }
        #pragma unroll
        for (int i = 0; i < 4; ++i) {
            int gk = k0 + brb + i;
            vb[i] = true;
            if (GUARD) { vb[i] = (gk < K); if (!vb[i]) gk = 0; }
            fb[i] = *(const float4*)(B + (size_t)gk * DFEAT + bc4 * 4);
        }
        if (GUARD) {                  // post-load select-zero (v_cndmask)
            if (!va) fa2 = make_float2(0.0f, 0.0f);
            #pragma unroll
            for (int i = 0; i < 4; ++i)
                if (!vb[i]) fb[i] = make_float4(0.0f, 0.0f, 0.0f, 0.0f);
        }

        // ---- Phase 2: convert + store to LDS (packed bf16 pairs)
        As_u[ar][ac >> 1] = pack_bf16(fa2.x, fa2.y);
        #pragma unroll
        for (int i = 0; i < 4; ++i) {
            const int col = bc4 * 4 + i;
            const float e0 = ((const float*)&fb[0])[i];
            const float e1 = ((const float*)&fb[1])[i];
            const float e2 = ((const float*)&fb[2])[i];
            const float e3 = ((const float*)&fb[3])[i];
            Bt_u[col][(brb >> 1) + 0] = pack_bf16(e0, e1);   // K-pairs packed
            Bt_u[col][(brb >> 1) + 1] = pack_bf16(e2, e3);
        }
        __syncthreads();

        // ---- Fragments as 16B LDS vector loads
        // A (16-bit 16x32 layout, ISA 7.12.2): lane half selects K-octet
        FragBF fa, fbv;
        const uint4* ap = (const uint4*)&As_u[l15][0];
        fa.q[0] = ap[hi];             // K = hi*8 .. hi*8+7
        fa.q[1] = ap[2 + hi];         // K = 16+hi*8 .. 23+hi*8
        // B (32x16): lane l15 = column, same K octets as A (hi, 2+hi)
        const uint4* bp = (const uint4*)&Bt_u[col0 + l15][0];
        fbv.q[0] = bp[hi];            // K = hi*8 .. hi*8+7
        fbv.q[1] = bp[2 + hi];        // K = 16+hi*8 .. 23+hi*8

        acc = __builtin_amdgcn_wmma_f32_16x16x32_bf16(
            /*neg_a=*/false, fa.v, /*neg_b=*/false, fbv.v,
            /*c_mod=*/(short)0, acc, /*reuse_a=*/false, /*reuse_b=*/false);
    }

    // ---- C layout (ISA 7.12.2): VGPR v -> M = v + 8*hi, N = l15
    const int d = col0 + l15;
    #pragma unroll
    for (int v = 0; v < 8; ++v) {
        const int n = row0 + v + 8 * hi;
        const size_t idx = (size_t)n * DFEAT + d;
        float g = acc[v];
        if (EPI) {
            g += bias[d];
            float bn = (conv[idx] - mean[d]) * invstd[d] * gmm[d] + bta[d];
            bn = bn > 0.0f ? bn : 0.0f;
            g = bn + g;
        }
        C[idx] = g;
    }
}

// ---------------------------------------------------------------------------
// Degree / normalization
// ---------------------------------------------------------------------------
__global__ __launch_bounds__(256)
void deg_init(float* deg) {           // self-loop contributes 1 to every node
    int n = blockIdx.x * 256 + threadIdx.x;
    if (n < N_NODES) deg[n] = 1.0f;
}

__global__ __launch_bounds__(256)
void deg_edges(const int* __restrict__ tgt, float* deg, int E) {
    int e = blockIdx.x * 256 + threadIdx.x;
    if (e < E) atomicAdd(&deg[tgt[e]], 1.0f);
}

__global__ __launch_bounds__(256)
void dinv_k(const float* __restrict__ deg, float* dinv) {
    int n = blockIdx.x * 256 + threadIdx.x;
    if (n < N_NODES) dinv[n] = rsqrtf(deg[n]);  // deg >= 1 (self loops)
}

// ---------------------------------------------------------------------------
// conv = b_gcn + dinv[n]^2 * hw[n]   (bias + self-loop term, non-atomic)
// ---------------------------------------------------------------------------
__global__ __launch_bounds__(256)
void conv_init(const float* __restrict__ hw, const float* __restrict__ dinv,
               const float* __restrict__ bg, float* __restrict__ conv)
{
    int i = blockIdx.x * 256 + threadIdx.x;         // over N*128/4 float4s
    int n  = i >> 5;
    int d4 = (i & 31) * 4;
    float di = dinv[n];
    float w  = di * di;
    float4 hv = ((const float4*)hw)[i];
    float4 o;
    o.x = bg[d4 + 0] + w * hv.x;
    o.y = bg[d4 + 1] + w * hv.y;
    o.z = bg[d4 + 2] + w * hv.z;
    o.w = bg[d4 + 3] + w * hv.w;
    ((float4*)conv)[i] = o;
}

// ---------------------------------------------------------------------------
// Edge scatter: one wave32 per edge; lane l owns features [4l, 4l+4).
// ---------------------------------------------------------------------------
__global__ __launch_bounds__(256)
void edge_scatter(const int* __restrict__ src, const int* __restrict__ tgt,
                  const float* __restrict__ dinv, const float* __restrict__ hw,
                  float* __restrict__ conv, int E)
{
    int e = blockIdx.x * 8 + (threadIdx.x >> 5);
    if (e >= E) return;
    int lane = threadIdx.x & 31;
    int s = src[e], t = tgt[e];
    float wgt = dinv[s] * dinv[t];
    float4 v = ((const float4*)(hw + (size_t)s * DFEAT))[lane];
    float* cp = conv + (size_t)t * DFEAT + lane * 4;
    atomicAdd(cp + 0, wgt * v.x);
    atomicAdd(cp + 1, wgt * v.y);
    atomicAdd(cp + 2, wgt * v.z);
    atomicAdd(cp + 3, wgt * v.w);
}

// ---------------------------------------------------------------------------
// BatchNorm statistics
// ---------------------------------------------------------------------------
__global__ void zero_stats(float* gsum, float* gsq) {
    int i = threadIdx.x;
    if (i < DFEAT) { gsum[i] = 0.0f; gsq[i] = 0.0f; }
}

__global__ __launch_bounds__(256)
void bn_partial(const float* __restrict__ conv, float* gsum, float* gsq)
{
    __shared__ float ssum[256], ssq[256];
    int d    = threadIdx.x & 127;
    int half = threadIdx.x >> 7;
    int rbase = blockIdx.x * 256;
    float s = 0.0f, q = 0.0f;
    for (int r = half; r < 256; r += 2) {
        float v = conv[(size_t)(rbase + r) * DFEAT + d];
        s += v; q += v * v;
    }
    ssum[threadIdx.x] = s; ssq[threadIdx.x] = q;
    __syncthreads();
    if (half == 0) {
        atomicAdd(&gsum[d], ssum[d] + ssum[d + 128]);
        atomicAdd(&gsq[d],  ssq[d]  + ssq[d + 128]);
    }
}

__global__ void bn_finalize(const float* gsum, const float* gsq,
                            float* mean, float* invstd)
{
    int d = threadIdx.x;
    if (d < DFEAT) {
        const float invN = 1.0f / (float)N_NODES;
        float m = gsum[d] * invN;
        float var = gsq[d] * invN - m * m;      // biased var (torch BN fwd)
        mean[d] = m;
        invstd[d] = rsqrtf(var + EPSBN);
    }
}

// ---------------------------------------------------------------------------
// Launch
// ---------------------------------------------------------------------------
extern "C" void kernel_launch(void* const* d_in, const int* in_sizes, int n_in,
                              void* d_out, int out_size, void* d_ws, size_t ws_size,
                              hipStream_t stream)
{
    const float* x      = (const float*)d_in[0];
    const int*   ei     = (const int*)d_in[1];
    const float* W_init = (const float*)d_in[3];
    const float* W_gcn  = (const float*)d_in[4];
    const float* b_gcn  = (const float*)d_in[5];
    const float* gamma  = (const float*)d_in[6];
    const float* beta   = (const float*)d_in[7];
    const float* W_res  = (const float*)d_in[8];
    const float* b_res  = (const float*)d_in[9];

    const int N = N_NODES, D = DFEAT, FIN = 74, L = 3;
    const int E = in_sizes[1] / 2;
    const int* src = ei;
    const int* tgt = ei + E;

    const size_t nodeBuf = (size_t)N * D * sizeof(float);       // 64 MiB
    size_t need = 3 * nodeBuf + 2 * (size_t)N * sizeof(float) + 4 * 512;
    if (ws_size < need) return;
    char* w = (char*)d_ws;
    float* h    = (float*)w; w += nodeBuf;
    float* hw   = (float*)w; w += nodeBuf;
    float* conv = (float*)w; w += nodeBuf;
    float* deg  = (float*)w; w += (size_t)N * sizeof(float);
    float* dinv = (float*)w; w += (size_t)N * sizeof(float);
    float* gsum = (float*)w; w += 512;
    float* gsq  = (float*)w; w += 512;
    float* mean = (float*)w; w += 512;
    float* istd = (float*)w; w += 512;

    // Degree + symmetric normalization (once)
    deg_init <<<N / 256, 256, 0, stream>>>(deg);
    deg_edges<<<(E + 255) / 256, 256, 0, stream>>>(tgt, deg, E);
    dinv_k   <<<N / 256, 256, 0, stream>>>(deg, dinv);

    // h = x @ W_init   (K = 74, zero-padded in LDS; clamped-load path)
    gemm_bf16_wmma<true, false><<<N / 16, 256, 0, stream>>>(x, W_init, h, FIN,
        nullptr, nullptr, nullptr, nullptr, nullptr, nullptr);

    for (int i = 0; i < L; ++i) {
        // hw = h @ W_gcn[i]
        gemm_bf16_wmma<false, false><<<N / 16, 256, 0, stream>>>(h,
            W_gcn + (size_t)i * D * D, hw, D,
            nullptr, nullptr, nullptr, nullptr, nullptr, nullptr);
        // conv = b_gcn[i] + dinv^2 * hw   (self-loop + bias)
        conv_init<<<N * D / 4 / 256, 256, 0, stream>>>(hw, dinv, b_gcn + i * D, conv);
        // conv[tgt] += dinv[src]*dinv[tgt] * hw[src]
        edge_scatter<<<(E + 7) / 8, 256, 0, stream>>>(src, tgt, dinv, hw, conv, E);
        // BN statistics
        zero_stats <<<1, 256, 0, stream>>>(gsum, gsq);
        bn_partial <<<N / 256, 256, 0, stream>>>(conv, gsum, gsq);
        bn_finalize<<<1, 128, 0, stream>>>(gsum, gsq, mean, istd);
        // h = relu(BN(conv)) + (h @ W_res[i] + b_res[i])
        // (in place over h is safe: each block reads only the A-rows it
        //  writes, and all its A reads complete before its stores)
        float* out = (i == L - 1) ? (float*)d_out : h;
        gemm_bf16_wmma<false, true><<<N / 16, 256, 0, stream>>>(h,
            W_res + (size_t)i * D * D, out, D,
            conv, mean, istd, gamma + i * D, beta + i * D, b_res + i * D);
    }
}
